// MultiHeadAttention_42356967473216
// MI455X (gfx1250) — matrix operand order repaired and checked by reading, not compiled, not run on hardware
//
#include <hip/hip_runtime.h>
#include <hip/hip_bf16.h>
#include <cstdint>

// ---------------- problem constants (from reference) ----------------
#define HN 16      // heads
#define DM 1024    // d_model
#define DH 64      // d_head
#define BB 4       // batch
#define SS 2048    // seq

typedef __bf16 bf16;
typedef __attribute__((ext_vector_type(16))) __bf16 v16bf;
typedef __attribute__((ext_vector_type(8)))  float  v8f;

// WMMA fragment layouts (wave32, CDNA5 ISA 7.12.2):
//  A (16x32, 16-bit): lane -> row M = lane&15, half = lane>>4,
//      element e -> K = 16*(e>>3) + 8*half + (e&7)
//  B (32x16, 16-bit): lane -> col N = lane&15, element e -> K = 16*half + e as this code packs it; A loads use the same order
//  C/D (16x16 f32):   vgpr r -> M = r + 8*half, N = lane&15
//
// "packed-A" order stores each 32-K chunk in natural K order (pack32 is identity),
// so an A fragment is lane-contiguous: a = *(v16bf*)(row + chunk + 16*half).

__device__ __forceinline__ int pack32(int k) {          // src K -> packed pos
    const int c = k >> 5, a = (k >> 3) & 3, w = k & 7;
    const int grp = a;
    return (c << 5) + (grp << 3) + w;
}
__device__ __forceinline__ int unpack32(int p) {        // packed pos -> src K
    const int c = p >> 5, grp = (p >> 3) & 3, w = p & 7;
    const int a = grp;
    return (c << 5) + (a << 3) + w;
}

// =====================================================================
// Prep 1: x (fp32 [B*S][DM]) -> xb (bf16, packed-A per 32-K chunk)
// =====================================================================
__global__ __launch_bounds__(256) void pack_x_kernel(
    const float* __restrict__ x, bf16* __restrict__ xb)
{
    const int idx = blockIdx.x * 256 + threadIdx.x;     // < 8'388'608
    const int row = idx >> 10, p = idx & 1023;
    xb[(size_t)row * DM + p] = (bf16)x[(size_t)row * DM + unpack32(p)];
}

// =====================================================================
// Prep 2: W_{Q,K,V} (fp32 [H][DM][DH]) -> bf16 B-fragment order
//   pW[h][c][n][half][e] = W[h][c*32 + 16*half + e][n],  c = K-chunk of 32
// =====================================================================
__global__ __launch_bounds__(256) void pack_wqkv_kernel(
    const float* __restrict__ Wq, const float* __restrict__ Wk,
    const float* __restrict__ Wv,
    bf16* __restrict__ pWq, bf16* __restrict__ pWk, bf16* __restrict__ pWv)
{
    const float* W; bf16* pW;
    if (blockIdx.z == 0)      { W = Wq; pW = pWq; }
    else if (blockIdx.z == 1) { W = Wk; pW = pWk; }
    else                      { W = Wv; pW = pWv; }
    const int o = blockIdx.x * 256 + threadIdx.x;       // < 1'048'576
    const int h = o >> 16, rem = o & 65535;
    const int c = (rem >> 11) & 31, n = (rem >> 5) & 63;
    const int half = (rem >> 4) & 1, e = rem & 15;
    const int kk = c * 32 + 16 * half + e;
    pW[o] = (bf16)W[((size_t)h * DM + kk) * DH + n];
}

// =====================================================================
// Prep 3: W_O (fp32 [DM][DM]) -> bf16 B-fragment order
//   pWo[c][n][half][e] = Wo[(c*32 + 16*half + e)][n]
// =====================================================================
__global__ __launch_bounds__(256) void pack_wo_kernel(
    const float* __restrict__ Wo, bf16* __restrict__ pWo)
{
    const int o = blockIdx.x * 256 + threadIdx.x;       // < 1'048'576
    const int c = o >> 15, rem = o & 32767;
    const int n = (rem >> 5) & 1023, half = (rem >> 4) & 1, e = rem & 15;
    const int kk = c * 32 + 16 * half + e;
    pWo[o] = (bf16)Wo[(size_t)kk * DM + n];
}

// =====================================================================
// Kernel 1: Q/K/V projection. grid = (64, H, 3), block = 256 (8 waves,
// one 16x64 tile each). All WMMA operands are single v16bf loads.
//   Q out: [b][h][s][64 packed-A]   K out: [b][h][s][64 natural]
//   V out: [b][h][64][S] (transposed for phase-3 B fragments)
// =====================================================================
__global__ __launch_bounds__(256) void qkv_kernel(
    const bf16* __restrict__ xb,
    const bf16* __restrict__ pWq, const bf16* __restrict__ pWk,
    const bf16* __restrict__ pWv,
    const float* __restrict__ bq, const float* __restrict__ bk,
    const float* __restrict__ bv,
    bf16* __restrict__ Qb, bf16* __restrict__ Kb, bf16* __restrict__ Vt)
{
    const int lane = threadIdx.x & 31;
    const int wave = threadIdx.x >> 5;
    const int m    = lane & 15;
    const int half = lane >> 4;
    const int rt   = blockIdx.x * 8 + wave;   // 0..511 row tiles over B*S
    const int h    = blockIdx.y;
    const int z    = blockIdx.z;

    const bf16* pW; const float* bias;
    if (z == 0)      { pW = pWq; bias = bq; }
    else if (z == 1) { pW = pWk; bias = bk; }
    else             { pW = pWv; bias = bv; }

    const int row0 = rt * 16;
    const bf16* xrow = xb + (size_t)(row0 + m) * DM;
    const bf16* pWh  = pW + (size_t)h * (32 * 64 * 32); // 32 chunks*64 n*32

    v8f acc[4] = {v8f{}, v8f{}, v8f{}, v8f{}};

    for (int k0 = 0; k0 < DM; k0 += 32) {
        const v16bf a = *(const v16bf*)(xrow + k0 + half * 16);
        const bf16* wchunk = pWh + (size_t)(k0 >> 5) * (64 * 32);
        #pragma unroll
        for (int nt = 0; nt < 4; ++nt) {
            const v16bf bm = *(const v16bf*)(wchunk + (nt * 16 + m) * 32 + half * 16);
            acc[nt] = __builtin_amdgcn_wmma_f32_16x16x32_bf16(
                false, a, false, bm, (short)0, acc[nt], false, false);
        }
    }

    #pragma unroll
    for (int nt = 0; nt < 4; ++nt) {
        #pragma unroll
        for (int r = 0; r < 8; ++r) {
            const int grow = row0 + r + 8 * half;      // row in [0, B*S)
            const int bb   = grow >> 11;               // / SS
            const int s    = grow & (SS - 1);
            const int e    = nt * 16 + m;
            const float val = acc[nt][r] + bias[h * DH + e];
            if (z == 0)
                Qb[(((size_t)bb * HN + h) * SS + s) * DH + pack32(e)] = (bf16)val;
            else if (z == 1)
                Kb[(((size_t)bb * HN + h) * SS + s) * DH + e] = (bf16)val;
            else
                Vt[(((size_t)bb * HN + h) * DH + e) * SS + s] = (bf16)val;
        }
    }
}

// =====================================================================
// Kernel 2: attention core. grid = (S/16, H, B), block = 256 (8 waves).
// 128 KB LDS fp32 score strip; probs written once (non-temporal).
// =====================================================================
__global__ __launch_bounds__(256) void attn_kernel(
    const bf16* __restrict__ Qb, const bf16* __restrict__ Kb,
    const bf16* __restrict__ Vt,
    float* __restrict__ probs, bf16* __restrict__ ctx)
{
    __shared__ float sc[16 * SS];       // score strip, reused for partials
    __shared__ float part[16][16];
    __shared__ float rowmax[16];
    __shared__ float rowsum[16];

    const int lane = threadIdx.x & 31;
    const int wave = threadIdx.x >> 5;
    const int m    = lane & 15;
    const int half = lane >> 4;
    const int qt   = blockIdx.x;
    const int h    = blockIdx.y;
    const int b    = blockIdx.z;
    const int q0   = qt * 16;
    const int L    = q0 + 16;           // causal valid-column count

    const bf16* Qh = Qb + ((size_t)b * HN + h) * SS * DH;
    const bf16* Kh = Kb + ((size_t)b * HN + h) * SS * DH;
    const bf16* Vh = Vt + ((size_t)b * HN + h) * DH * SS;
    float* prow = probs + (((size_t)b * HN + h) * SS + q0) * SS;

    // ---- Phase 1: scores = (Q . K^T)/8 for column tiles c0 <= q0 ----
    const bf16* qp = Qh + (size_t)(q0 + m) * DH;
    const v16bf aq0 = *(const v16bf*)(qp + half * 16);        // K-dim 0..31
    const v16bf aq1 = *(const v16bf*)(qp + 32 + half * 16);   // K-dim 32..63

    for (int ct = wave; ct <= qt; ct += 8) {
        const int c0 = ct * 16;
        const bf16* kp = Kh + (size_t)(c0 + m) * DH;
        const v16bf b0 = *(const v16bf*)(kp + half * 16);
        const v16bf b1 = *(const v16bf*)(kp + 32 + half * 16);
        v8f c = {};
        c = __builtin_amdgcn_wmma_f32_16x16x32_bf16(false, aq0, false, b0, (short)0, c, false, false);
        c = __builtin_amdgcn_wmma_f32_16x16x32_bf16(false, aq1, false, b1, (short)0, c, false, false);
        #pragma unroll
        for (int r = 0; r < 8; ++r) {
            const int qrow = q0 + r + 8 * half;
            const int col  = c0 + m;
            float v = c[r] * 0.125f;                    // 1/sqrt(64)
            if (col > qrow) v = -__builtin_inff();      // diagonal-tile mask
            sc[(r + 8 * half) * SS + col] = v;
        }
    }
    __syncthreads();

    // ---- Phase 2: row softmax over [0, L); single probs write (NT) ----
    const int r = threadIdx.x >> 4;
    const int j = threadIdx.x & 15;
    float mx = -__builtin_inff();
    for (int c2 = j; c2 < L; c2 += 16) mx = fmaxf(mx, sc[r * SS + c2]);
    part[r][j] = mx;
    __syncthreads();
    if (j == 0) {
        float v = part[r][0];
        #pragma unroll
        for (int t = 1; t < 16; ++t) v = fmaxf(v, part[r][t]);
        rowmax[r] = v;
    }
    __syncthreads();
    const float rm = rowmax[r];
    float sum = 0.f;
    for (int c2 = j; c2 < L; c2 += 16) {
        const float p = __expf(sc[r * SS + c2] - rm);
        sc[r * SS + c2] = p;
        sum += p;
    }
    part[r][j] = sum;
    __syncthreads();
    if (j == 0) {
        float v = 0.f;
        #pragma unroll
        for (int t = 0; t < 16; ++t) v += part[r][t];
        rowsum[r] = v;
    }
    __syncthreads();
    const float inv = 1.0f / rowsum[r];
    for (int c2 = j; c2 < L; c2 += 16) {
        const float p = sc[r * SS + c2] * inv;
        sc[r * SS + c2] = p;
        __builtin_nontemporal_store(p, &prow[(size_t)r * SS + c2]);
    }
    for (int c2 = L + j; c2 < SS; c2 += 16)             // masked tail == 0
        __builtin_nontemporal_store(0.f, &prow[(size_t)r * SS + c2]);
    if (L & 31)                                          // pad to K-chunk 32
        sc[r * SS + L + j] = 0.f;
    __syncthreads();

    // ---- Phase 3: ctx = probs(16xL) . V(Lx64), K split across waves ----
    const int nch = (L + 31) >> 5;
    v8f acc[4] = {v8f{}, v8f{}, v8f{}, v8f{}};
    for (int ch = wave; ch < nch; ch += 8) {
        const int kbase = ch * 32;
        v16bf a;
        #pragma unroll
        for (int e = 0; e < 16; ++e) {
            const int kk = kbase + (half << 4) + e;
            a[e] = (bf16)sc[m * SS + kk];
        }
        #pragma unroll
        for (int nt = 0; nt < 4; ++nt) {
            const v16bf bm = *(const v16bf*)
                (Vh + (size_t)(nt * 16 + m) * SS + kbase + half * 16);
            acc[nt] = __builtin_amdgcn_wmma_f32_16x16x32_bf16(
                false, a, false, bm, (short)0, acc[nt], false, false);
        }
    }
    __syncthreads();                     // all waves done reading sc
    float* partial = sc;                 // reuse: [8 waves][16 rows][64]
    #pragma unroll
    for (int nt = 0; nt < 4; ++nt)
        #pragma unroll
        for (int rr = 0; rr < 8; ++rr)
            partial[((wave * 16) + rr + 8 * half) * 64 + nt * 16 + m] = acc[nt][rr];
    __syncthreads();
    for (int idx = threadIdx.x; idx < 16 * 64; idx += 256) {
        const int rr = idx >> 6;
        const int e  = idx & 63;
        float s = 0.f;
        #pragma unroll
        for (int w = 0; w < 8; ++w) s += partial[(w * 16 + rr) * 64 + e];
        // ctx bf16 [b][s][1024], packed-A order for the output GEMM
        ctx[((size_t)b * SS + q0 + rr) * DM + pack32(h * DH + e)] = (bf16)s;
    }
}

// =====================================================================
// Kernel 3: out = ctx[8192x1024] . W_O[1024x1024] + b_O (fp32 out).
// grid = (64, 16 col-tiles of 64), block = 256. v16bf operand loads.
// =====================================================================
__global__ __launch_bounds__(256) void outproj_kernel(
    const bf16* __restrict__ ctx, const bf16* __restrict__ pWo,
    const float* __restrict__ bo, float* __restrict__ out)
{
    const int lane = threadIdx.x & 31;
    const int wave = threadIdx.x >> 5;
    const int m    = lane & 15;
    const int half = lane >> 4;
    const int rt   = blockIdx.x * 8 + wave;   // 0..511
    const int col0 = blockIdx.y * 64;
    const int row0 = rt * 16;

    const bf16* crow = ctx + (size_t)(row0 + m) * DM;
    v8f acc[4] = {v8f{}, v8f{}, v8f{}, v8f{}};

    for (int k0 = 0; k0 < DM; k0 += 32) {
        const v16bf a = *(const v16bf*)(crow + k0 + half * 16);
        const bf16* wchunk = pWo + (size_t)(k0 >> 5) * (DM * 32);
        #pragma unroll
        for (int nt = 0; nt < 4; ++nt) {
            const v16bf bm = *(const v16bf*)
                (wchunk + (size_t)(col0 + nt * 16 + m) * 32 + half * 16);
            acc[nt] = __builtin_amdgcn_wmma_f32_16x16x32_bf16(
                false, a, false, bm, (short)0, acc[nt], false, false);
        }
    }

    #pragma unroll
    for (int nt = 0; nt < 4; ++nt) {
        #pragma unroll
        for (int r = 0; r < 8; ++r) {
            const int grow = row0 + r + 8 * half;
            const int col  = col0 + nt * 16 + m;
            out[(size_t)grow * DM + col] = acc[nt][r] + bo[col];
        }
    }
}

// =====================================================================
extern "C" void kernel_launch(void* const* d_in, const int* in_sizes, int n_in,
                              void* d_out, int out_size, void* d_ws, size_t ws_size,
                              hipStream_t stream) {
    const float* x  = (const float*)d_in[0];
    const float* Wq = (const float*)d_in[1];
    const float* Wk = (const float*)d_in[2];
    const float* Wv = (const float*)d_in[3];
    const float* Wo = (const float*)d_in[4];
    const float* bq = (const float*)d_in[5];
    const float* bk = (const float*)d_in[6];
    const float* bv = (const float*)d_in[7];
    const float* bo = (const float*)d_in[8];

    float* probs = (float*)d_out;                              // [B,H,S,S]
    float* out   = probs + (size_t)BB * HN * SS * SS;          // [B,S,DM]

    const size_t xe  = (size_t)BB * SS * DM;                   // 8.39M
    const size_t we  = (size_t)HN * DM * DH;                   // 1.05M
    const size_t qe  = (size_t)BB * HN * SS * DH;              // 8.39M

    bf16* xb  = (bf16*)d_ws;
    bf16* pWq = xb  + xe;
    bf16* pWk = pWq + we;
    bf16* pWv = pWk + we;
    bf16* pWo = pWv + we;
    bf16* Qb  = pWo + (size_t)DM * DM;
    bf16* Kb  = Qb  + qe;
    bf16* Vt  = Kb  + qe;
    bf16* Cb  = Vt  + qe;                                      // ctx [B,S,DM]

    pack_x_kernel  <<<dim3((unsigned)(xe / 256)), 256, 0, stream>>>(x, xb);
    pack_wqkv_kernel<<<dim3((unsigned)(we / 256), 1, 3), 256, 0, stream>>>(
        Wq, Wk, Wv, pWq, pWk, pWv);
    pack_wo_kernel <<<dim3((unsigned)((size_t)DM * DM / 256)), 256, 0, stream>>>(Wo, pWo);

    qkv_kernel    <<<dim3(64, HN, 3), 256, 0, stream>>>(
        xb, pWq, pWk, pWv, bq, bk, bv, Qb, Kb, Vt);
    attn_kernel   <<<dim3(SS / 16, HN, BB), 256, 0, stream>>>(Qb, Kb, Vt, probs, Cb);
    outproj_kernel<<<dim3(64, DM / 64), 256, 0, stream>>>(Cb, pWo, bo, out);
}
